// MoEST_HER2_75926431858740
// MI455X (gfx1250) — matrix operand order repaired and checked by reading, not compiled, not run on hardware
//
#include <hip/hip_runtime.h>
#include <hip/hip_bf16.h>
#include <math.h>

// ---------------------------------------------------------------------------
// MoEST_HER2 forward, CDNA5 (gfx1250) WMMA implementation.
// All large GEMMs -> v_wmma_f32_16x16x32_bf16 (f32 accumulate).
// Weights are converted fp32 -> bf16 *transposed* (W^T [N,K]) so both the
// A tile and the B tile stream through LDS as contiguous b128 ops with no
// per-element guards in the K loop.
// ---------------------------------------------------------------------------

typedef __bf16 v16bf __attribute__((ext_vector_type(16)));
typedef float  v8f   __attribute__((ext_vector_type(8)));

struct __align__(16) U4 { unsigned x, y, z, w; };

__device__ __forceinline__ __bf16 f2b(float f) {
  unsigned u = __builtin_bit_cast(unsigned, f);
  unsigned r = (u + 0x7FFFu + ((u >> 16) & 1u)) >> 16;   // round-to-nearest-even
  return __builtin_bit_cast(__bf16, (unsigned short)r);
}
__device__ __forceinline__ float gelu_f(float x) {
  return 0.5f * x * (1.0f + erff(x * 0.70710678118654752440f));   // exact erf GELU
}
__device__ __forceinline__ float softplus_f(float x) {
  return (x > 20.0f) ? x : log1pf(expf(x));
}

// ------------------------------- generic WMMA GEMM --------------------------
// C_out = epilogue(A[MxK](bf16) @ W (given as W^T[N,K] bf16) + bias)
// modes: 0 store f32 (+opt bf16 mirror)
//        1 GELU, store f32 (opt) + bf16 mirror (opt)
//        2 C += rowscale[r] * val            (MoE top-1 combine)
//        4 C += gelu(val)                    (fuse pos-encoder branch into z)
//        3 gene split: even col -> mu = softplus(v)*lib[r]+1e-6
//                      odd  col -> theta = softplus(v)+1e-6
#define BM 128
#define BN 128
#define BK 32
#define LDSTR 40   // BK + 8 bf16 pad (80B row stride -> conflict-free frag reads)

struct GemmP {
  const __bf16* A; const __bf16* Wt; const float* bias;
  float* C; __bf16* Cb;
  const float* rowscale; const float* lib;
  float* mu; float* th;
  int M, N, K, G2, mode;
};

__global__ void __launch_bounds__(256) wmma_gemm(GemmP p) {
  __shared__ __bf16 sA[BM * LDSTR];
  __shared__ __bf16 sB[BN * LDSTR];     // sB[col][k] (W^T rows are columns of W)

  const int tid     = threadIdx.x;
  const int rowBase = blockIdx.y * BM;
  const int colBase = blockIdx.x * BN;
  const int wave = tid >> 5, lane = tid & 31;
  const int wm = wave >> 1, wn = wave & 1;       // 4x2 wave grid, each 32(M)x64(N)
  const int fr = lane & 15;
  const int hi = lane >> 4;                      // lane half
  const int ka = hi ? 8 : 0;                     // A frag: K {ka..ka+7, ka+16..ka+23}
  const int kb = hi ? 8 : 0;                     // B frag: K {kb..kb+7, kb+16..kb+23}

  union Frag { v16bf v; U4 q[2]; };

  const v8f vzero = {0.f,0.f,0.f,0.f,0.f,0.f,0.f,0.f};
  v8f acc[2][4] = {{vzero,vzero,vzero,vzero},{vzero,vzero,vzero,vzero}};

  // tile-load mapping (identical for A and B): 128 rows x 32 k, 16 bf16/thread
  const int ldr  = tid >> 1;              // 0..127
  const int lseg = (tid & 1) * 16;        // 0 or 16
  const int bcol = (colBase + ldr < p.N) ? (colBase + ldr) : (p.N - 1);  // clamp
  const __bf16* gA = p.A  + (size_t)(rowBase + ldr) * p.K + lseg;
  const __bf16* gB = p.Wt + (size_t)bcol            * p.K + lseg;
  __bf16* lA = sA + ldr * LDSTR + lseg;
  __bf16* lB = sB + ldr * LDSTR + lseg;

  for (int k0 = 0; k0 < p.K; k0 += BK) {
    __syncthreads();
    {   // A tile 128x32 and B tile 128(cols)x32: two b128 each, no branches
      U4 a0 = *(const U4*)(gA + k0);
      U4 a1 = *(const U4*)(gA + k0 + 8);
      U4 b0 = *(const U4*)(gB + k0);
      U4 b1 = *(const U4*)(gB + k0 + 8);
      *(U4*)(lA)     = a0;
      *(U4*)(lA + 8) = a1;
      *(U4*)(lB)     = b0;
      *(U4*)(lB + 8) = b1;
      if (k0 + BK < p.K) {
        __builtin_prefetch(gA + k0 + BK, 0, 0);
        __builtin_prefetch(gB + k0 + BK, 0, 0);
      }
    }
    __syncthreads();

    Frag a[2], b[4];
    #pragma unroll
    for (int i = 0; i < 2; ++i) {
      const __bf16* pa = sA + (wm * 32 + i * 16 + fr) * LDSTR;
      a[i].q[0] = *(const U4*)(pa + ka);
      a[i].q[1] = *(const U4*)(pa + ka + 16);
    }
    #pragma unroll
    for (int j = 0; j < 4; ++j) {
      const __bf16* pb = sB + (wn * 64 + j * 16 + fr) * LDSTR;
      b[j].q[0] = *(const U4*)(pb + kb);
      b[j].q[1] = *(const U4*)(pb + kb + 16);
    }
    #pragma unroll
    for (int i = 0; i < 2; ++i)
      #pragma unroll
      for (int j = 0; j < 4; ++j)
        acc[i][j] = __builtin_amdgcn_wmma_f32_16x16x32_bf16(
            false, a[i].v, false, b[j].v, (short)0, acc[i][j], false, false);
  }

  // epilogue: C/D layout -> VGPR v: (M = v + 8*hi, N = fr) within each 16x16 tile
  #pragma unroll
  for (int i = 0; i < 2; ++i) {
    #pragma unroll
    for (int j = 0; j < 4; ++j) {
      #pragma unroll
      for (int v = 0; v < 8; ++v) {
        int r = rowBase + wm * 32 + i * 16 + v + (hi ? 8 : 0);
        int c = colBase + wn * 64 + j * 16 + fr;
        if (c >= p.N) continue;
        float val = acc[i][j][v];
        if (p.bias) val += p.bias[c];
        size_t o = (size_t)r * p.N + c;
        switch (p.mode) {
          case 0:
            p.C[o] = val;
            if (p.Cb) p.Cb[o] = f2b(val);
            break;
          case 1: {
            float g = gelu_f(val);
            if (p.C)  p.C[o]  = g;
            if (p.Cb) p.Cb[o] = f2b(g);
          } break;
          case 2:
            p.C[o] += p.rowscale[r] * val;
            break;
          case 4:
            p.C[o] += gelu_f(val);
            break;
          case 3: {
            float s = softplus_f(val);
            size_t og = (size_t)r * p.G2 + (c >> 1);
            if ((c & 1) == 0) p.mu[og] = s * p.lib[r] + 1e-6f;
            else              p.th[og] = s + 1e-6f;
          } break;
        }
      }
    }
  }
}

// ------------------------------- small kernels ------------------------------

__global__ void k_f2b(const float* __restrict__ s, __bf16* __restrict__ d, int n) {
  int i = blockIdx.x * blockDim.x + threadIdx.x;
  if (i < n) d[i] = f2b(s[i]);
}

// fp32 [K,N] -> bf16 transposed [N,K]
__global__ void k_f2bT(const float* __restrict__ s, __bf16* __restrict__ d,
                       int K, int N) {
  int idx = blockIdx.x * blockDim.x + threadIdx.x;
  if (idx >= N * K) return;
  int n = idx / K, k = idx - n * K;
  d[idx] = f2b(s[(size_t)k * N + n]);
}

// Fourier features: four = [sin(2pi * pos@B), cos(...)], stored bf16 [B,256]
__global__ void k_fourier(const float* __restrict__ pos, const float* __restrict__ fB,
                          __bf16* __restrict__ four, int Bn) {
  int idx = blockIdx.x * blockDim.x + threadIdx.x;
  if (idx >= Bn * 128) return;
  int b = idx >> 7, j = idx & 127;
  float x = 6.28318530717958647692f *
            (pos[b*3+0]*fB[j] + pos[b*3+1]*fB[128+j] + pos[b*3+2]*fB[256+j]);
  four[(size_t)b*256 + j]       = f2b(sinf(x));
  four[(size_t)b*256 + 128 + j] = f2b(cosf(x));
}

// router: logits = [z, grad]@rW + rb; softmax; top-1 gate stored as gateT[e*B+b]
__global__ void __launch_bounds__(256) k_router(const float* __restrict__ z,
    const float* __restrict__ grad, const float* __restrict__ rW,
    const float* __restrict__ rb, float* __restrict__ gateT, int Bn) {
  int row  = (blockIdx.x * blockDim.x + threadIdx.x) >> 5;
  int lane = threadIdx.x & 31;
  if (row >= Bn) return;
  const float* zr = z + (size_t)row * 256;
  float a0 = 0.f, a1 = 0.f, a2 = 0.f, a3 = 0.f;
  for (int k = lane; k < 256; k += 32) {
    float zv = zr[k];
    const float* w = rW + k * 4;
    a0 += zv * w[0]; a1 += zv * w[1]; a2 += zv * w[2]; a3 += zv * w[3];
  }
  for (int off = 16; off > 0; off >>= 1) {
    a0 += __shfl_down(a0, off); a1 += __shfl_down(a1, off);
    a2 += __shfl_down(a2, off); a3 += __shfl_down(a3, off);
  }
  if (lane == 0) {
    float g = grad[row];
    float l[4] = { a0 + g*rW[256*4+0] + rb[0], a1 + g*rW[256*4+1] + rb[1],
                   a2 + g*rW[256*4+2] + rb[2], a3 + g*rW[256*4+3] + rb[3] };
    float m = l[0]; for (int e = 1; e < 4; ++e) m = fmaxf(m, l[e]);
    float pr[4]; float s = 0.f;
    for (int e = 0; e < 4; ++e) { pr[e] = expf(l[e] - m); s += pr[e]; }
    int am = 0; for (int e = 1; e < 4; ++e) if (pr[e] > pr[am]) am = e;
    for (int e = 0; e < 4; ++e) gateT[(size_t)e*Bn + row] = (e == am) ? pr[e]/s : 0.f;
  }
}

// LayerNorm(256) + GELU -> bf16; one 256-thread block per row
__global__ void __launch_bounds__(256) k_ln(const float* __restrict__ t,
    const float* __restrict__ g, const float* __restrict__ beta,
    __bf16* __restrict__ out) {
  __shared__ float red[256];
  int b = blockIdx.x, j = threadIdx.x;
  float x = t[(size_t)b*256 + j];
  red[j] = x; __syncthreads();
  for (int s = 128; s > 0; s >>= 1) { if (j < s) red[j] += red[j + s]; __syncthreads(); }
  float mu = red[0] * (1.0f/256.0f); __syncthreads();
  float d = x - mu;
  red[j] = d * d; __syncthreads();
  for (int s = 128; s > 0; s >>= 1) { if (j < s) red[j] += red[j + s]; __syncthreads(); }
  float var = red[0] * (1.0f/256.0f);
  float y = gelu_f(g[j] * d * rsqrtf(var + 1e-5f) + beta[j]);
  out[(size_t)b*256 + j] = f2b(y);
}

// func head: sigmoid(gelu(z@W1+b1)@W2+b2); one 64-thread block per row
__global__ void __launch_bounds__(64) k_func(const float* __restrict__ zf,
    const float* __restrict__ W1, const float* __restrict__ b1,
    const float* __restrict__ W2, const float* __restrict__ b2,
    float* __restrict__ out) {
  __shared__ float h[64];
  int b = blockIdx.x, j = threadIdx.x;
  const float* zr = zf + (size_t)b*256;
  float a = b1[j];
  for (int k = 0; k < 256; ++k) a += zr[k] * W1[k*64 + j];
  h[j] = gelu_f(a) * W2[j];
  __syncthreads();
  for (int s = 32; s > 0; s >>= 1) { if (j < s) h[j] += h[j + s]; __syncthreads(); }
  if (j == 0) out[b] = 1.0f / (1.0f + expf(-(h[0] + b2[0])));
}

// align head final: [B,128]@[128,30]+b; one thread per output
__global__ void k_align(const float* __restrict__ a1, const float* __restrict__ W2,
                        const float* __restrict__ b2, float* __restrict__ out, int Bn) {
  int idx = blockIdx.x * blockDim.x + threadIdx.x;
  if (idx >= Bn * 30) return;
  int b = idx / 30, j = idx - b * 30;
  const float* ar = a1 + (size_t)b * 128;
  float a = b2[j];
  for (int k = 0; k < 128; ++k) a += ar[k] * W2[k*30 + j];
  out[idx] = a;
}

// ------------------------------- launcher -----------------------------------

static inline void gemm(hipStream_t st, const __bf16* A, const __bf16* Wt, const float* bias,
                        float* C, __bf16* Cb, const float* rs, const float* lib,
                        float* mu, float* th, int M, int N, int K, int G2, int mode) {
  GemmP p{A, Wt, bias, C, Cb, rs, lib, mu, th, M, N, K, G2, mode};
  dim3 grid((N + BN - 1) / BN, M / BM);
  wmma_gemm<<<grid, dim3(256), 0, st>>>(p);
}

extern "C" void kernel_launch(void* const* d_in, const int* in_sizes, int n_in,
                              void* d_out, int out_size, void* d_ws, size_t ws_size,
                              hipStream_t stream) {
  const int DVIS = 1024, DH = 256, E = 4, G = 5000, SCVI = 30;
  const int Bn = in_sizes[0] / DVIS;              // 16384

  const float* vis    = (const float*)d_in[0];
  const float* pos    = (const float*)d_in[1];
  const float* grad   = (const float*)d_in[2];
  const float* lib    = (const float*)d_in[3];
  const float* fB     = (const float*)d_in[4];
  const float* img_W  = (const float*)d_in[5];
  const float* img_b  = (const float*)d_in[6];
  const float* pos_W  = (const float*)d_in[7];
  const float* pos_b  = (const float*)d_in[8];
  const float* rout_W = (const float*)d_in[9];
  const float* rout_b = (const float*)d_in[10];
  const float* e_W1   = (const float*)d_in[11];
  const float* e_b1   = (const float*)d_in[12];
  const float* e_W2   = (const float*)d_in[13];
  const float* e_b2   = (const float*)d_in[14];
  const float* gd_W1  = (const float*)d_in[15];
  const float* gd_b1  = (const float*)d_in[16];
  const float* gd_g   = (const float*)d_in[17];
  const float* gd_be  = (const float*)d_in[18];
  const float* gd_W2  = (const float*)d_in[19];
  const float* gd_b2  = (const float*)d_in[20];
  const float* ap_W1  = (const float*)d_in[21];
  const float* ap_b1  = (const float*)d_in[22];
  const float* ap_W2  = (const float*)d_in[23];
  const float* ap_b2  = (const float*)d_in[24];
  const float* fh_W1  = (const float*)d_in[25];
  const float* fh_b1  = (const float*)d_in[26];
  const float* fh_W2  = (const float*)d_in[27];
  const float* fh_b2  = (const float*)d_in[28];

  float* out   = (float*)d_out;
  float* o_mu  = out;                               // [B,G]
  float* o_th  = out + (size_t)Bn * G;              // [B,G]
  float* o_fn  = out + (size_t)2 * Bn * G;          // [B,1]
  float* o_al  = o_fn + Bn;                         // [B,SCVI]

  // -------- workspace carve-out --------
  size_t off = 0;
  auto take = [&](size_t bytes) -> char* {
    char* pp = (char*)d_ws + off;
    off += (bytes + 255) & ~(size_t)255;
    return pp;
  };
  __bf16* vis_b   = (__bf16*)take((size_t)Bn * DVIS * 2);
  __bf16* imgWt   = (__bf16*)take((size_t)DVIS * DH * 2);        // [DH][DVIS]
  __bf16* posWt   = (__bf16*)take((size_t)2 * 128 * DH * 2);     // [DH][256]
  __bf16* eW1t    = (__bf16*)take((size_t)E * DH * 4 * DH * 2);  // E x [4DH][DH]
  __bf16* eW2t    = (__bf16*)take((size_t)E * 4 * DH * DH * 2);  // E x [DH][4DH]
  __bf16* gdW1t   = (__bf16*)take((size_t)DH * DH * 2);          // [DH][DH]
  __bf16* gdW2t   = (__bf16*)take((size_t)DH * 2 * G * 2);       // [2G][DH]
  __bf16* apW1t   = (__bf16*)take((size_t)DH * 128 * 2);         // [128][DH]
  __bf16* four_b  = (__bf16*)take((size_t)Bn * 256 * 2);
  float*  zf      = (float*) take((size_t)Bn * DH * 4);      // z, later z_final
  __bf16* zb      = (__bf16*)take((size_t)Bn * DH * 2);      // bf16 snapshot of z
  __bf16* zfb     = (__bf16*)take((size_t)Bn * DH * 2);      // bf16 of z_final
  float*  gate    = (float*) take((size_t)E * Bn * 4);       // [E][B]
  __bf16* Hb      = (__bf16*)take((size_t)Bn * 4 * DH * 2);  // expert hidden
  float*  tr      = (float*) take((size_t)Bn * DH * 4);      // pre-LN t
  __bf16* tb      = (__bf16*)take((size_t)Bn * DH * 2);      // post-LN/GELU t
  float*  a1f     = (float*) take((size_t)Bn * 128 * 4);     // align hidden

  auto cvt = [&](const float* s, __bf16* d, int n) {
    k_f2b<<<dim3((n + 255) / 256), dim3(256), 0, stream>>>(s, d, n);
  };
  auto cvtT = [&](const float* s, __bf16* d, int K, int N) {
    k_f2bT<<<dim3((N * K + 255) / 256), dim3(256), 0, stream>>>(s, d, K, N);
  };

  // -------- bf16 casts: activations row-major, weights transposed ----------
  cvt (vis,   vis_b,  Bn * DVIS);
  cvtT(img_W, imgWt,  DVIS, DH);
  cvtT(pos_W, posWt,  2 * 128, DH);
  for (int e = 0; e < E; ++e) {
    cvtT(e_W1 + (size_t)e * DH * 4 * DH, eW1t + (size_t)e * DH * 4 * DH, DH, 4 * DH);
    cvtT(e_W2 + (size_t)e * 4 * DH * DH, eW2t + (size_t)e * 4 * DH * DH, 4 * DH, DH);
  }
  cvtT(gd_W1, gdW1t, DH, DH);
  cvtT(gd_W2, gdW2t, DH, 2 * G);
  cvtT(ap_W1, apW1t, DH, 128);

  // -------- encoders: z = vis@img_W + img_b + gelu(four@pos_W + pos_b) -----
  k_fourier<<<dim3((Bn * 128 + 255) / 256), dim3(256), 0, stream>>>(pos, fB, four_b, Bn);
  gemm(stream, vis_b,  imgWt, img_b, zf, nullptr, nullptr, nullptr, nullptr, nullptr,
       Bn, DH, DVIS, 0, /*store*/0);
  gemm(stream, four_b, posWt, pos_b, zf, nullptr, nullptr, nullptr, nullptr, nullptr,
       Bn, DH, 2 * 128, 0, /*+=gelu*/4);

  // -------- router (top-1 gate) + bf16 snapshot of z -----------------------
  k_router<<<dim3((Bn * 32 + 255) / 256), dim3(256), 0, stream>>>(zf, grad, rout_W, rout_b, gate, Bn);
  cvt(zf, zb, Bn * DH);

  // -------- dense MoE (reference computes all experts), gated accumulate ---
  for (int e = 0; e < E; ++e) {
    gemm(stream, zb, eW1t + (size_t)e * DH * 4 * DH, e_b1 + (size_t)e * 4 * DH,
         nullptr, Hb, nullptr, nullptr, nullptr, nullptr,
         Bn, 4 * DH, DH, 0, /*gelu->bf16*/1);
    gemm(stream, Hb, eW2t + (size_t)e * 4 * DH * DH, e_b2 + (size_t)e * DH,
         zf, nullptr, gate + (size_t)e * Bn, nullptr, nullptr, nullptr,
         Bn, DH, 4 * DH, 0, /*+=gate*val*/2);
  }
  cvt(zf, zfb, Bn * DH);   // zf is now z_final

  // -------- gene decoder: GEMM -> LN+GELU -> big GEMM with mu/theta split --
  gemm(stream, zfb, gdW1t, gd_b1, tr, nullptr, nullptr, nullptr, nullptr, nullptr,
       Bn, DH, DH, 0, /*store*/0);
  k_ln<<<dim3(Bn), dim3(256), 0, stream>>>(tr, gd_g, gd_be, tb);
  gemm(stream, tb, gdW2t, gd_b2, nullptr, nullptr, nullptr, lib, o_mu, o_th,
       Bn, 2 * G, DH, G, /*gene split*/3);

  // -------- align projector ------------------------------------------------
  gemm(stream, zfb, apW1t, ap_b1, a1f, nullptr, nullptr, nullptr, nullptr, nullptr,
       Bn, 128, DH, 0, /*gelu*/1);
  k_align<<<dim3((Bn * SCVI + 255) / 256), dim3(256), 0, stream>>>(a1f, ap_W2, ap_b2, o_al, Bn);

  // -------- func head ------------------------------------------------------
  k_func<<<dim3(Bn), dim3(64), 0, stream>>>(zf, fh_W1, fh_b1, fh_W2, fh_b2, o_fn);
}
